// NeighbourhoodAttnBlock_44538810860341
// MI455X (gfx1250) — matrix operand order repaired and checked by reading, not compiled, not run on hardware
//
#include <hip/hip_runtime.h>
#include <hip/hip_bf16.h>
#include <math.h>

// ---------------------------------------------------------------------------
// NeighbourhoodAttnBlock for MI455X (gfx1250), bf16 WMMA pipeline.
//   B=2, H=W=64, D=512, NH=8, DH=64, KERNEL=7
// Stages:
//   1) f32 -> bf16 convert (x, w_qkv, w_out)
//   2) QKV GEMM (WMMA bf16, async global->LDS double-buffered) -> q,k,v
//   3) neighbourhood attention per 8x8 query tile (WMMA bf16 twice + softmax)
//   4) output projection GEMM (WMMA bf16, async global->LDS) -> f32 d_out
// ---------------------------------------------------------------------------

typedef __bf16 bf16;
typedef __bf16 v16bf __attribute__((ext_vector_type(16)));
typedef float  v8f   __attribute__((ext_vector_type(8)));

union Frag16 { v16bf v; uint4 u[2]; };

static __device__ __forceinline__ v8f wmma_bf16(const Frag16& a, const Frag16& b, v8f c) {
  // D(16x16 f32) = A(16x32 bf16) * B(32x16 bf16) + C
  return __builtin_amdgcn_wmma_f32_16x16x32_bf16(false, a.v, false, b.v, (short)0, c, false, false);
}

// Async copy of 32 contiguous bytes global -> LDS (two B128 ops, ASYNCcnt +2).
// INST_OFFSET is added to BOTH the global and LDS addresses, so the same
// address registers serve both 16-byte chunks.
static __device__ __forceinline__ void async_copy32(const bf16* gptr, bf16* lptr) {
  unsigned l = (unsigned)(uintptr_t)lptr;                // LDS address: low 32 bits of the pointer
  unsigned long long g = (unsigned long long)(uintptr_t)gptr;
  asm volatile("global_load_async_to_lds_b128 %0, %1, off\n\t"
               "global_load_async_to_lds_b128 %0, %1, off offset:16"
               :: "v"(l), "v"(g) : "memory");
}

// ---------------------------------------------------------------------------
// f32 -> bf16 convert, 4 elements per thread
// ---------------------------------------------------------------------------
struct bf16x4 { bf16 a, b, c, d; };

__global__ __launch_bounds__(256) void cvt_bf16_kernel(const float4* __restrict__ src,
                                                       bf16x4* __restrict__ dst, int n4) {
  int i = blockIdx.x * blockDim.x + threadIdx.x;
  if (i >= n4) return;
  float4 f = src[i];
  bf16x4 o;
  o.a = (bf16)f.x; o.b = (bf16)f.y; o.c = (bf16)f.z; o.d = (bf16)f.w;
  dst[i] = o;
}

// ---------------------------------------------------------------------------
// Shared 128x128 tile GEMM core: C = A[M x K] * Bt[N x K]^T, K multiple of 32.
// 256 threads = 8 waves in a 2(M) x 4(N) grid; each wave owns 64x32 (4x2 frags).
// Global->LDS staging uses async-to-LDS with 2 LDS buffers: while tile i is
// consumed by WMMA, tile i+1 is in flight on the async pipe (ASYNCcnt).
// As/Bs are double buffers: 2 x 128x32 bf16 each.
// ---------------------------------------------------------------------------
__device__ __forceinline__ void gemm_tile_128x128(const bf16* __restrict__ A,
                                                  const bf16* __restrict__ Bt,
                                                  int K, int m0, int n0,
                                                  v8f (&acc)[4][2],
                                                  bf16* As, bf16* Bs) {
  const int tid  = threadIdx.x;
  const int lane = tid & 31, wid = tid >> 5;
  const int half = lane >> 4, l16 = lane & 15;
  const int wm = wid >> 2, wn = wid & 3;

  const v8f ZV = {0.f, 0.f, 0.f, 0.f, 0.f, 0.f, 0.f, 0.f};
#pragma unroll
  for (int mi = 0; mi < 4; ++mi)
#pragma unroll
    for (int ni = 0; ni < 2; ++ni) acc[mi][ni] = ZV;

  const int r = tid >> 1;          // LDS row 0..127 (each loaded by 2 threads)
  const int c = (tid & 1) << 4;    // element offset 0 or 16 within BK=32
  const bf16* gA = A + (size_t)(m0 + r) * K + c;
  const bf16* gB = Bt + (size_t)(n0 + r) * K + c;
  bf16* lA = As + r * 32 + c;
  bf16* lB = Bs + r * 32 + c;

  // prologue: stage tile 0 into buffer 0 (4 async ops per thread)
  async_copy32(gA, lA);
  async_copy32(gB, lB);

  for (int k0 = 0; k0 < K; k0 += 32) {
    const int buf = (k0 >> 5) & 1;
    if (k0 + 32 < K) {
      // stage next tile into the other buffer (safe: it was last read in the
      // previous iteration, which ended with a workgroup barrier)
      async_copy32(gA + k0 + 32, lA + (buf ^ 1) * 4096);
      async_copy32(gB + k0 + 32, lB + (buf ^ 1) * 4096);
      // retire current tile's 4 ops; leave next tile's 4 in flight
      asm volatile("s_wait_asynccnt 0x4" ::: "memory");
    } else {
      asm volatile("s_wait_asynccnt 0x0" ::: "memory");
    }
    __syncthreads();

    const bf16* Ab = As + buf * 4096;
    const bf16* Bb = Bs + buf * 4096;

    // ---- fragments + WMMA ----
    Frag16 af[4], bfr[2];
#pragma unroll
    for (int mi = 0; mi < 4; ++mi) {
      int m = wm * 64 + mi * 16 + l16;
      af[mi].u[0] = *reinterpret_cast<const uint4*>(Ab + m * 32 + 16 * half);
      af[mi].u[1] = *reinterpret_cast<const uint4*>(Ab + m * 32 + 16 * half + 8);
    }
#pragma unroll
    for (int ni = 0; ni < 2; ++ni) {
      int n = wn * 32 + ni * 16 + l16;
      const uint4* p = reinterpret_cast<const uint4*>(Bb + n * 32 + 16 * half);
      bfr[ni].u[0] = p[0]; bfr[ni].u[1] = p[1];
    }
#pragma unroll
    for (int mi = 0; mi < 4; ++mi)
#pragma unroll
      for (int ni = 0; ni < 2; ++ni)
        acc[mi][ni] = wmma_bf16(af[mi], bfr[ni], acc[mi][ni]);
    __syncthreads();
  }
}

// ---------------------------------------------------------------------------
// Stage 2: QKV projection. A = x_bf16 [8192 x 512], Bt = w_qkv [1536 x 512].
// Output column gn decomposes as (t, nh, e) = (gn/512, (gn%512)/64, gn%64).
// ---------------------------------------------------------------------------
__global__ __launch_bounds__(256) void qkv_gemm_kernel(const bf16* __restrict__ A,
                                                       const bf16* __restrict__ Bt,
                                                       bf16* __restrict__ qb,
                                                       bf16* __restrict__ kb,
                                                       bf16* __restrict__ vb) {
  __shared__ __attribute__((aligned(16))) bf16 As[2 * 128 * 32];
  __shared__ __attribute__((aligned(16))) bf16 Bs[2 * 128 * 32];
  v8f acc[4][2];
  const int m0 = blockIdx.x * 128, n0 = blockIdx.y * 128;
  gemm_tile_128x128(A, Bt, 512, m0, n0, acc, As, Bs);

  const int tid = threadIdx.x, lane = tid & 31, wid = tid >> 5;
  const int half = lane >> 4, l16 = lane & 15;
  const int wm = wid >> 2, wn = wid & 3;
#pragma unroll
  for (int mi = 0; mi < 4; ++mi)
#pragma unroll
    for (int ni = 0; ni < 2; ++ni)
#pragma unroll
      for (int r = 0; r < 8; ++r) {
        int gm = m0 + wm * 64 + mi * 16 + r + 8 * half;
        int gn = n0 + wn * 32 + ni * 16 + l16;
        float val = acc[mi][ni][r];
        int t = gn >> 9, rem = gn & 511;
        int nh = rem >> 6, e = rem & 63;
        int bb = gm >> 12, s = gm & 4095;
        size_t dst = (((size_t)(bb * 8 + nh)) * 4096 + s) * 64 + e;
        bf16 hv = (bf16)val;
        if (t == 0)      qb[dst] = hv;
        else if (t == 1) kb[dst] = hv;
        else             vb[dst] = hv;
      }
}

// ---------------------------------------------------------------------------
// Stage 4: output projection. A = attn_out [8192 x 512], Bt = w_out [512 x 512].
// ---------------------------------------------------------------------------
__global__ __launch_bounds__(256) void out_gemm_kernel(const bf16* __restrict__ A,
                                                       const bf16* __restrict__ Bt,
                                                       float* __restrict__ out) {
  __shared__ __attribute__((aligned(16))) bf16 As[2 * 128 * 32];
  __shared__ __attribute__((aligned(16))) bf16 Bs[2 * 128 * 32];
  v8f acc[4][2];
  const int m0 = blockIdx.x * 128, n0 = blockIdx.y * 128;
  gemm_tile_128x128(A, Bt, 512, m0, n0, acc, As, Bs);

  const int tid = threadIdx.x, lane = tid & 31, wid = tid >> 5;
  const int half = lane >> 4, l16 = lane & 15;
  const int wm = wid >> 2, wn = wid & 3;
#pragma unroll
  for (int mi = 0; mi < 4; ++mi)
#pragma unroll
    for (int ni = 0; ni < 2; ++ni)
#pragma unroll
      for (int r = 0; r < 8; ++r) {
        int gm = m0 + wm * 64 + mi * 16 + r + 8 * half;
        int gn = n0 + wn * 32 + ni * 16 + l16;
        out[(size_t)gm * 512 + gn] = acc[mi][ni][r];
      }
}

// ---------------------------------------------------------------------------
// Stage 3: neighbourhood attention.
// One block per (b, nh, 8x8 query tile). Key window: fixed clamped 14x14
// (=196 keys, covers the union of every query's clamped 7x7 window),
// padded to 224 (7 WMMA N-frags). Exact per-(q,k) mask applied to scores.
// ---------------------------------------------------------------------------
__global__ __launch_bounds__(256) void attn_kernel(const bf16* __restrict__ qg,
                                                   const bf16* __restrict__ kg,
                                                   const bf16* __restrict__ vg,
                                                   bf16* __restrict__ ab) {
  __shared__ __attribute__((aligned(16))) bf16 KP[224 * 64];  // K rows, later P rows
  __shared__ __attribute__((aligned(16))) bf16 Vt[64 * 224];  // V transposed [d][key]
  __shared__ float redmax[64 * 2];
  __shared__ float redsum[64 * 2];

  const int tid = threadIdx.x, lane = tid & 31, wid = tid >> 5;
  const int half = lane >> 4, l16 = lane & 15;

  const int tile = blockIdx.x;
  const int bnh = tile >> 6;                 // (b*8 + nh)
  const int t64 = tile & 63;
  const int h0 = (t64 >> 3) << 3, w0 = (t64 & 7) << 3;
  const int kh0 = min(max(h0 - 3, 0), 50);
  const int kw0 = min(max(w0 - 3, 0), 50);

  const bf16* qbase = qg + (size_t)bnh * 4096 * 64;
  const bf16* kbase = kg + (size_t)bnh * 4096 * 64;
  const bf16* vbase = vg + (size_t)bnh * 4096 * 64;

  // ---- load K tile row-major, V tile transposed; zero the 196..223 padding ----
#pragma unroll
  for (int it = 0; it < 7; ++it) {
    int chunk = tid + it * 256;              // 0..1791 = 224 rows * 8 chunks
    int row = chunk >> 3;                    // key index 0..223
    int c8  = (chunk & 7) << 3;              // head-dim offset 0,8,..,56
    uint4 kv = {0, 0, 0, 0}, vv = {0, 0, 0, 0};
    if (row < 196) {
      int srow = (kh0 + row / 14) * 64 + (kw0 + row % 14);
      kv = *reinterpret_cast<const uint4*>(kbase + (size_t)srow * 64 + c8);
      vv = *reinterpret_cast<const uint4*>(vbase + (size_t)srow * 64 + c8);
    }
    *reinterpret_cast<uint4*>(&KP[row * 64 + c8]) = kv;
    const bf16* vp = reinterpret_cast<const bf16*>(&vv);
#pragma unroll
    for (int j = 0; j < 8; ++j) Vt[(c8 + j) * 224 + row] = vp[j];
  }
  __syncthreads();

  // ---- scores S = Q * K^T : 64 x 224, wave grid 4(M:16 rows) x 2(N:112 cols) ----
  const int wm = wid >> 1;
  const int wn = wid & 1;
  const v8f ZV = {0.f, 0.f, 0.f, 0.f, 0.f, 0.f, 0.f, 0.f};
  v8f sc[7];
#pragma unroll
  for (int f = 0; f < 7; ++f) sc[f] = ZV;

#pragma unroll
  for (int ks = 0; ks < 2; ++ks) {           // head dim 64 = 2 x K32
    Frag16 a;
    int m = wm * 16 + l16;                   // query index 0..63
    int qs = (h0 + (m >> 3)) * 64 + (w0 + (m & 7));
    const bf16* qr = qbase + (size_t)qs * 64 + ks * 32;
    a.u[0] = *reinterpret_cast<const uint4*>(qr + 16 * half);
    a.u[1] = *reinterpret_cast<const uint4*>(qr + 16 * half + 8);
#pragma unroll
    for (int f = 0; f < 7; ++f) {
      Frag16 bb;
      int n = wn * 112 + f * 16 + l16;       // key index
      const uint4* p = reinterpret_cast<const uint4*>(&KP[n * 64 + ks * 32 + 16 * half]);
      bb.u[0] = p[0]; bb.u[1] = p[1];
      sc[f] = wmma_bf16(a, bb, sc[f]);
    }
  }

  // ---- scale + exact neighbourhood mask + row max ----
  const float NEG = -__builtin_inff();
#pragma unroll
  for (int r = 0; r < 8; ++r) {
    int q = wm * 16 + r + 8 * half;
    int qh = h0 + (q >> 3), qw = w0 + (q & 7);
    int sh = min(max(qh - 3, 0), 57), sw = min(max(qw - 3, 0), 57);
    float mx = NEG;
#pragma unroll
    for (int f = 0; f < 7; ++f) {
      int col = wn * 112 + f * 16 + l16;
      int kh = kh0 + col / 14, kw = kw0 + col % 14;
      bool valid = (col < 196) && (kh >= sh) && (kh < sh + 7) && (kw >= sw) && (kw < sw + 7);
      float v = valid ? sc[f][r] * 0.125f : NEG;   // 0.125 = DH^-0.5
      sc[f][r] = v;
      mx = fmaxf(mx, v);
    }
#pragma unroll
    for (int d = 1; d < 16; d <<= 1) mx = fmaxf(mx, __shfl_xor(mx, d, 32));
    if (l16 == 0) redmax[q * 2 + wn] = mx;
  }
  __syncthreads();

  // ---- exp + row sum ----
#pragma unroll
  for (int r = 0; r < 8; ++r) {
    int q = wm * 16 + r + 8 * half;
    float M = fmaxf(redmax[q * 2], redmax[q * 2 + 1]);
    float sum = 0.f;
#pragma unroll
    for (int f = 0; f < 7; ++f) {
      float p = __expf(sc[f][r] - M);        // masked -> exp(-inf) = 0
      sc[f][r] = p;
      sum += p;
    }
#pragma unroll
    for (int d = 1; d < 16; d <<= 1) sum += __shfl_xor(sum, d, 32);
    if (l16 == 0) redsum[q * 2 + wn] = sum;
  }
  __syncthreads();

  // ---- normalize, write P (bf16) over the retired K region: P[q][key] ----
#pragma unroll
  for (int r = 0; r < 8; ++r) {
    int q = wm * 16 + r + 8 * half;
    float inv = 1.f / (redsum[q * 2] + redsum[q * 2 + 1]);
#pragma unroll
    for (int f = 0; f < 7; ++f) {
      int col = wn * 112 + f * 16 + l16;
      KP[q * 224 + col] = (bf16)(sc[f][r] * inv);
    }
  }
  __syncthreads();

  // ---- O = P * V : 64 x 64, wave grid 4(M:16 rows) x 2(N:32 cols) ----
  const int wm2 = wid >> 1, wn2 = wid & 1;
  v8f oacc[2];
  oacc[0] = ZV; oacc[1] = ZV;
#pragma unroll
  for (int ks = 0; ks < 7; ++ks) {           // 224 keys = 7 x K32
    Frag16 a;
    int m = wm2 * 16 + l16;
    const bf16* pr = &KP[m * 224 + ks * 32];
    a.u[0] = *reinterpret_cast<const uint4*>(pr + 16 * half);
    a.u[1] = *reinterpret_cast<const uint4*>(pr + 16 * half + 8);
#pragma unroll
    for (int ni = 0; ni < 2; ++ni) {
      Frag16 bb;
      int dcol = wn2 * 32 + ni * 16 + l16;
      const uint4* p = reinterpret_cast<const uint4*>(&Vt[dcol * 224 + ks * 32 + 16 * half]);
      bb.u[0] = p[0]; bb.u[1] = p[1];
      oacc[ni] = wmma_bf16(a, bb, oacc[ni]);
    }
  }

  // ---- write attention output, channel = nh*64 + d ----
  const int b = bnh >> 3, nh = bnh & 7;
#pragma unroll
  for (int ni = 0; ni < 2; ++ni)
#pragma unroll
    for (int r = 0; r < 8; ++r) {
      int q = wm2 * 16 + r + 8 * half;
      int dcol = wn2 * 32 + ni * 16 + l16;
      int srow = (h0 + (q >> 3)) * 64 + (w0 + (q & 7));
      ab[((size_t)(b * 4096 + srow)) * 512 + nh * 64 + dcol] = (bf16)oacc[ni][r];
    }
}

// ---------------------------------------------------------------------------
// Host launcher
// ---------------------------------------------------------------------------
extern "C" void kernel_launch(void* const* d_in, const int* in_sizes, int n_in,
                              void* d_out, int out_size, void* d_ws, size_t ws_size,
                              hipStream_t stream) {
  (void)in_sizes; (void)n_in; (void)out_size; (void)ws_size;
  const float* x    = (const float*)d_in[0];  // [2,64,64,512]
  const float* wqkv = (const float*)d_in[1];  // [1536,512]
  const float* wout = (const float*)d_in[2];  // [512,512]
  float* out = (float*)d_out;                 // [2,64,64,512] f32

  char* ws = (char*)d_ws;
  size_t off = 0;
  auto take = [&](size_t bytes) {
    void* p = ws + off;
    off += (bytes + 255) & ~(size_t)255;
    return p;
  };
  const size_t NX = (size_t)8192 * 512;       // b*s x d
  bf16* xb    = (bf16*)take(NX * 2);
  bf16* wqkvb = (bf16*)take((size_t)1536 * 512 * 2);
  bf16* woutb = (bf16*)take((size_t)512 * 512 * 2);
  bf16* qb    = (bf16*)take(NX * 2);          // [b][nh][s][64]
  bf16* kb    = (bf16*)take(NX * 2);
  bf16* vb    = (bf16*)take(NX * 2);
  bf16* abuf  = (bf16*)take(NX * 2);          // attention output [b*s][512]

  cvt_bf16_kernel<<<4096, 256, 0, stream>>>((const float4*)x, (bf16x4*)xb, (int)(NX / 4));
  cvt_bf16_kernel<<<768, 256, 0, stream>>>((const float4*)wqkv, (bf16x4*)wqkvb, 1536 * 512 / 4);
  cvt_bf16_kernel<<<256, 256, 0, stream>>>((const float4*)wout, (bf16x4*)woutb, 512 * 512 / 4);

  qkv_gemm_kernel<<<dim3(64, 12), 256, 0, stream>>>(xb, wqkvb, qb, kb, vb);
  attn_kernel<<<1024, 256, 0, stream>>>(qb, kb, vb, abuf);
  out_gemm_kernel<<<dim3(64, 4), 256, 0, stream>>>(abuf, woutb, out);
}
